// Attn_24404004176217
// MI455X (gfx1250) — matrix operand order repaired and checked by reading, not compiled, not run on hardware
//
#include <hip/hip_runtime.h>

// ---------------------------------------------------------------------------
// CDNA5 (gfx1250) attention block:
//   x@Wqkv -> causal QK^T -> softmax -> PV -> @Wproj
// All matmuls: v_wmma_f32_16x16x32_bf16 (bf16 in, f32 accumulate).
// All GEMM tile staging: global_load_async_to_lds_b128 (ASYNCcnt DMA),
// double-buffered so DMA of tile k+1 overlaps WMMA on tile k.
// ---------------------------------------------------------------------------

typedef __bf16 bf16_t;
typedef __bf16 v16bf __attribute__((ext_vector_type(16)));
typedef float  v8f   __attribute__((ext_vector_type(8)));

#define CDIM 1024
#define TSEQ 2048
#define NBAT 8
#define MROWS (NBAT * TSEQ)   // 16384

#define TILE_M 128
#define TILE_N 128
#define TILE_K 32
#define LDSS   40   // LDS row stride in bf16 (32 data + 8 pad -> 80B rows, 16B aligned)

// ---- LDS byte-offset of a shared pointer (ISA 10.2: flat addr[31:0] = LDS addr)
__device__ __forceinline__ uint32_t lds_u32(const void* p) {
  return (uint32_t)(uintptr_t)p;
}

// ---- async global->LDS 16B copy, GVS mode (saddr base + per-lane u32 offset)
__device__ __forceinline__ void async_b128(uint32_t lds_addr, const void* base,
                                           uint32_t voff) {
  asm volatile("global_load_async_to_lds_b128 %0, %1, %2"
               :: "v"(lds_addr), "v"(voff), "s"((unsigned long long)(uintptr_t)base)
               : "memory");
}

// ---- WMMA fragment loaders (ISA 7.12.2 layouts, wave32) --------------------
// A 16x32 bf16: M = lane&15; e=0..7 -> K = hi*8+e ; e=8..15 -> K = 16+hi*8+(e-8)
__device__ __forceinline__ v16bf frag_ldA(const bf16_t* base, int rowBase, int lane) {
  const int l16 = lane & 15, hi = lane >> 4;
  const bf16_t* row = base + (rowBase + l16) * LDSS;
  union { v16bf v; uint4 q[2]; } u;
  u.q[0] = *(const uint4*)(row + hi * 8);
  u.q[1] = *(const uint4*)(row + 16 + hi * 8);
  return u.v;
}
// B 32x16 bf16 stored as LDS[n][k]: N = lane&15; e=0..7 -> K = hi*8+e ; e=8..15 -> K = 16+hi*8+(e-8)
__device__ __forceinline__ v16bf frag_ldB(const bf16_t* base, int rowBase, int lane) {
  const int l16 = lane & 15, hi = lane >> 4;
  const bf16_t* row = base + (rowBase + l16) * LDSS;
  union { v16bf v; uint4 q[2]; } u;
  u.q[0] = *(const uint4*)(row + hi * 8);
  u.q[1] = *(const uint4*)(row + 16 + hi * 8);
  return u.v;
}

// ---------------------------------------------------------------------------
// bf16 GEMM with pre-transposed B:  C[M,N] = A[M,K] * Bt[N,K]^T
// A row-major [M][K] bf16, Bt row-major [N][K] bf16 (i.e. already the [n][k]
// layout the B fragments want).  256 threads = 8 waves (4 M x 2 N),
// block tile 128x128, K step 32, double-buffered async LDS staging.
// ---------------------------------------------------------------------------
template<bool OUT_BF16, bool OUT_TRANS, bool CAUSALK>
__global__ __launch_bounds__(256) void gemm_bf16_async(
    const bf16_t* __restrict__ A, const bf16_t* __restrict__ Bt,
    void* __restrict__ Cdst,
    int K, int lda, int ldbt, int ldc,
    long long strideA, long long strideB, long long strideC)
{
  __shared__ __align__(16) bf16_t sA[2][TILE_M * LDSS];
  __shared__ __align__(16) bf16_t sB[2][TILE_N * LDSS];

  const int tid  = threadIdx.x;
  const int lane = tid & 31;
  const int wid  = tid >> 5;
  const int wm   = wid & 3;              // 4 waves along M
  const int wn   = wid >> 2;             // 2 waves along N
  const int m0   = blockIdx.x * TILE_M;
  const int n0   = blockIdx.y * TILE_N;
  const long long bz = blockIdx.z;

  int Keff = K;
  if constexpr (CAUSALK) {
    const int lim = m0 + TILE_M;
    Keff = (lim < K) ? lim : K;
  }
  const int nk = Keff / TILE_K;

  const bf16_t* Ab = A  + bz * strideA + (size_t)m0 * lda;
  const bf16_t* Bb = Bt + bz * strideB + (size_t)n0 * ldbt;

  // per-thread 16B chunks: 512 per tile -> 2 per thread; r = row, c = 16B col
  const int ch_r[2] = { (tid + 0)   >> 2, (tid + 256) >> 2 };
  const int ch_c[2] = { (tid + 0)   & 3,  (tid + 256) & 3  };

  auto issue_tile = [&](int k0, int buf) {
    const uint32_t aBase = lds_u32(&sA[buf][0]);
    const uint32_t bBase = lds_u32(&sB[buf][0]);
#pragma unroll
    for (int i = 0; i < 2; ++i) {
      const int r = ch_r[i], c = ch_c[i];
      async_b128(aBase + (uint32_t)(r * (LDSS * 2) + c * 16), Ab,
                 (uint32_t)(((size_t)r * lda + k0 + c * 8) * 2));
      async_b128(bBase + (uint32_t)(r * (LDSS * 2) + c * 16), Bb,
                 (uint32_t)(((size_t)r * ldbt + k0 + c * 8) * 2));
    }
  };

  const v8f vzero = {0.f, 0.f, 0.f, 0.f, 0.f, 0.f, 0.f, 0.f};
  v8f acc[2][4];
#pragma unroll
  for (int i = 0; i < 2; ++i)
#pragma unroll
    for (int j = 0; j < 4; ++j) acc[i][j] = vzero;

  issue_tile(0, 0);
  for (int kt = 0; kt < nk; ++kt) {
    if (kt + 1 < nk) {
      issue_tile((kt + 1) * TILE_K, (kt + 1) & 1);
      asm volatile("s_wait_asynccnt 4" ::: "memory");   // tile kt landed
    } else {
      asm volatile("s_wait_asynccnt 0" ::: "memory");
    }
    __syncthreads();

    const bf16_t* cA = sA[kt & 1];
    const bf16_t* cB = sB[kt & 1];
    v16bf af[2], bfr[4];
    af[0] = frag_ldA(cA, wm * 32, lane);
    af[1] = frag_ldA(cA, wm * 32 + 16, lane);
#pragma unroll
    for (int fn = 0; fn < 4; ++fn)
      bfr[fn] = frag_ldB(cB, wn * 64 + fn * 16, lane);
#pragma unroll
    for (int fm = 0; fm < 2; ++fm)
#pragma unroll
      for (int fn = 0; fn < 4; ++fn)
        acc[fm][fn] = __builtin_amdgcn_wmma_f32_16x16x32_bf16(
            false, af[fm], false, bfr[fn], (short)0, acc[fm][fn], false, false);
    __syncthreads();   // all waves done reading before this buf is re-filled
  }

  // ---- store (C/D layout: VGPR r -> M = base + hi*8 + r, N = lane16) ----
  const int l16 = lane & 15, hi = lane >> 4;
#pragma unroll
  for (int fm = 0; fm < 2; ++fm)
#pragma unroll
    for (int fn = 0; fn < 4; ++fn) {
      const int mb = m0 + wm * 32 + fm * 16 + hi * 8;
      const int nc = n0 + wn * 64 + fn * 16 + l16;
#pragma unroll
      for (int r = 0; r < 8; ++r) {
        const float v = acc[fm][fn][r];
        if constexpr (OUT_TRANS) {        // C[nc][m]  (bf16, e.g. V^T)
          ((bf16_t*)Cdst + bz * strideC)[(size_t)nc * ldc + (mb + r)] = (bf16_t)v;
        } else if constexpr (OUT_BF16) {
          ((bf16_t*)Cdst + bz * strideC)[(size_t)(mb + r) * ldc + nc] = (bf16_t)v;
        } else {
          ((float*)Cdst + bz * strideC)[(size_t)(mb + r) * ldc + nc] = v;
        }
      }
    }
}

// ---------------------------------------------------------------------------
// S[b,i,j] = (Q[b,i,:] . K[b,j,:]) / sqrt(C), causal masked, bf16 out.
// Q and K both [row][chan] bf16 -> both tiles staged with async B128 copies.
// Upper-triangle blocks (kb > qb) skipped entirely (never read downstream).
// ---------------------------------------------------------------------------
__global__ __launch_bounds__(256) void attn_scores_kernel(
    const bf16_t* __restrict__ Qg, const bf16_t* __restrict__ Kg,
    bf16_t* __restrict__ S)
{
  const int qb = blockIdx.x, kb = blockIdx.y, b = blockIdx.z;
  if (kb > qb) return;

  __shared__ __align__(16) bf16_t sQ[2][TILE_M * LDSS];
  __shared__ __align__(16) bf16_t sK[2][TILE_N * LDSS];

  const int tid = threadIdx.x, lane = tid & 31, wid = tid >> 5;
  const int wm = wid & 3, wn = wid >> 2;
  const int m0 = qb * TILE_M, n0 = kb * TILE_N;

  const bf16_t* Qb = Qg + (size_t)b * TSEQ * CDIM + (size_t)m0 * CDIM;
  const bf16_t* Kb = Kg + (size_t)b * TSEQ * CDIM + (size_t)n0 * CDIM;

  const int ch_r[2] = { (tid + 0) >> 2, (tid + 256) >> 2 };
  const int ch_c[2] = { (tid + 0) & 3,  (tid + 256) & 3  };

  auto issue_tile = [&](int k0, int buf) {
    const uint32_t qBase = lds_u32(&sQ[buf][0]);
    const uint32_t kBase = lds_u32(&sK[buf][0]);
#pragma unroll
    for (int i = 0; i < 2; ++i) {
      const int r = ch_r[i], c = ch_c[i];
      async_b128(qBase + (uint32_t)(r * (LDSS * 2) + c * 16), Qb,
                 (uint32_t)(((size_t)r * CDIM + k0 + c * 8) * 2));
      async_b128(kBase + (uint32_t)(r * (LDSS * 2) + c * 16), Kb,
                 (uint32_t)(((size_t)r * CDIM + k0 + c * 8) * 2));
    }
  };

  const v8f vzero = {0.f, 0.f, 0.f, 0.f, 0.f, 0.f, 0.f, 0.f};
  v8f acc[2][4];
#pragma unroll
  for (int i = 0; i < 2; ++i)
#pragma unroll
    for (int j = 0; j < 4; ++j) acc[i][j] = vzero;

  const int nk = CDIM / TILE_K;
  issue_tile(0, 0);
  for (int kt = 0; kt < nk; ++kt) {
    if (kt + 1 < nk) {
      issue_tile((kt + 1) * TILE_K, (kt + 1) & 1);
      asm volatile("s_wait_asynccnt 4" ::: "memory");
    } else {
      asm volatile("s_wait_asynccnt 0" ::: "memory");
    }
    __syncthreads();

    const bf16_t* cQ = sQ[kt & 1];
    const bf16_t* cK = sK[kt & 1];
    v16bf af[2], bfr[4];
    af[0] = frag_ldA(cQ, wm * 32, lane);
    af[1] = frag_ldA(cQ, wm * 32 + 16, lane);
#pragma unroll
    for (int fn = 0; fn < 4; ++fn)
      bfr[fn] = frag_ldB(cK, wn * 64 + fn * 16, lane);
#pragma unroll
    for (int fm = 0; fm < 2; ++fm)
#pragma unroll
      for (int fn = 0; fn < 4; ++fn)
        acc[fm][fn] = __builtin_amdgcn_wmma_f32_16x16x32_bf16(
            false, af[fm], false, bfr[fn], (short)0, acc[fm][fn], false, false);
    __syncthreads();
  }

  const float scale = 0.03125f;  // 1/sqrt(1024)
  const int l16 = lane & 15, hi = lane >> 4;
  bf16_t* Sb = S + (size_t)b * TSEQ * TSEQ;
#pragma unroll
  for (int fm = 0; fm < 2; ++fm)
#pragma unroll
    for (int fn = 0; fn < 4; ++fn) {
      const int mb = m0 + wm * 32 + fm * 16 + hi * 8;
      const int nc = n0 + wn * 64 + fn * 16 + l16;
#pragma unroll
      for (int r = 0; r < 8; ++r) {
        const int row = mb + r;
        const float v = (nc <= row) ? acc[fm][fn][r] * scale : -1e30f;
        Sb[(size_t)row * TSEQ + nc] = (bf16_t)v;
      }
    }
}

// ---------------------------------------------------------------------------
// In-place causal softmax: one wave32 per row; rewrites S as P = exp(s-m)/l.
// Masked entries (-1e30) exp() to 0; whole 128-wide block span rewritten so
// the PV GEMM reads it blindly.
// ---------------------------------------------------------------------------
__global__ __launch_bounds__(256) void softmax_kernel(bf16_t* __restrict__ S)
{
  const int lane = threadIdx.x & 31;
  const int row  = blockIdx.x * 8 + (threadIdx.x >> 5);
  const int b    = blockIdx.y;
  bf16_t* srow = S + ((size_t)b * TSEQ + row) * TSEQ;

  const int ncols  = row + 1;
  const int ncolsW = ((row >> 7) + 1) << 7;

  float m = -3.0e38f;
  for (int c = lane; c < ncols; c += 32) m = fmaxf(m, (float)srow[c]);
#pragma unroll
  for (int off = 16; off > 0; off >>= 1) m = fmaxf(m, __shfl_xor(m, off, 32));

  float l = 0.f;
  for (int c = lane; c < ncols; c += 32) l += __expf((float)srow[c] - m);
#pragma unroll
  for (int off = 16; off > 0; off >>= 1) l += __shfl_xor(l, off, 32);

  const float inv = 1.0f / l;
  for (int c = lane; c < ncolsW; c += 32)
    srow[c] = (bf16_t)(__expf((float)srow[c] - m) * inv);
}

// ---------------------------------------------------------------------------
// Precision/layout prep: f32 -> bf16 elementwise, and f32 -> bf16 transpose.
// ---------------------------------------------------------------------------
__global__ __launch_bounds__(256) void convert_bf16_kernel(
    const float* __restrict__ in, bf16_t* __restrict__ out)
{
  const size_t i = ((size_t)blockIdx.x * 256 + threadIdx.x) * 4;
  const float4 d = *(const float4*)(in + i);
  bf16_t t[4] = {(bf16_t)d.x, (bf16_t)d.y, (bf16_t)d.z, (bf16_t)d.w};
  *(uint2*)(out + i) = *(const uint2*)t;
}

// in [1024][cols] f32 -> out [cols][1024] bf16 (coalesced writes)
__global__ __launch_bounds__(256) void transpose_bf16_kernel(
    const float* __restrict__ in, bf16_t* __restrict__ out, int cols)
{
  const size_t o = (size_t)blockIdx.x * 256 + threadIdx.x;
  const int r  = (int)(o & 1023);      // input row (K)
  const int cc = (int)(o >> 10);       // input col (N)
  out[o] = (bf16_t)in[(size_t)r * cols + cc];
}

// ---------------------------------------------------------------------------
extern "C" void kernel_launch(void* const* d_in, const int* in_sizes, int n_in,
                              void* d_out, int out_size, void* d_ws, size_t ws_size,
                              hipStream_t stream) {
  (void)in_sizes; (void)n_in; (void)out_size; (void)ws_size;

  const float* x     = (const float*)d_in[0];   // [B,T,C]  f32
  const float* Wqkv  = (const float*)d_in[1];   // [C,3C]   f32
  const float* Wproj = (const float*)d_in[2];   // [C,C]    f32
  float* out = (float*)d_out;                   // [B,T,C]  f32

  // workspace layout (bf16): xb 32MB | Q 32MB | K 32MB | Vt 32MB | S 64MB |
  //                          O 32MB | Wqkv_t 6MB | Wproj_t 2MB  = 232MB
  char* ws = (char*)d_ws;
  size_t ofs = 0;
  bf16_t* xb   = (bf16_t*)(ws + ofs); ofs += (size_t)MROWS * CDIM * 2;
  bf16_t* Qb   = (bf16_t*)(ws + ofs); ofs += (size_t)MROWS * CDIM * 2;
  bf16_t* Kb   = (bf16_t*)(ws + ofs); ofs += (size_t)MROWS * CDIM * 2;
  bf16_t* Vt   = (bf16_t*)(ws + ofs); ofs += (size_t)MROWS * CDIM * 2;  // [C][B*T]
  bf16_t* S    = (bf16_t*)(ws + ofs); ofs += (size_t)NBAT * TSEQ * TSEQ * 2;
  bf16_t* Ob   = (bf16_t*)(ws + ofs); ofs += (size_t)MROWS * CDIM * 2;
  bf16_t* Wqt  = (bf16_t*)(ws + ofs); ofs += (size_t)3 * CDIM * CDIM * 2; // [3C][C]
  bf16_t* Wpt  = (bf16_t*)(ws + ofs);                                     // [C][C]

  const dim3 blk(256);

  // 0) precision/layout prep
  convert_bf16_kernel<<<dim3((MROWS * CDIM) / (256 * 4)), blk, 0, stream>>>(x, xb);
  transpose_bf16_kernel<<<dim3((3 * CDIM * CDIM) / 256), blk, 0, stream>>>(
      Wqkv, Wqt, 3 * CDIM);
  transpose_bf16_kernel<<<dim3((CDIM * CDIM) / 256), blk, 0, stream>>>(
      Wproj, Wpt, CDIM);

  // 1) QKV projection (three N-slices of Wqkv_t; V stored transposed)
  gemm_bf16_async<true, false, false>
      <<<dim3(MROWS / TILE_M, CDIM / TILE_N, 1), blk, 0, stream>>>(
          xb, Wqt, Qb, CDIM, CDIM, CDIM, CDIM, 0, 0, 0);
  gemm_bf16_async<true, false, false>
      <<<dim3(MROWS / TILE_M, CDIM / TILE_N, 1), blk, 0, stream>>>(
          xb, Wqt + (size_t)CDIM * CDIM, Kb, CDIM, CDIM, CDIM, CDIM, 0, 0, 0);
  gemm_bf16_async<true, true, false>   // V^T: [C][B*T]
      <<<dim3(MROWS / TILE_M, CDIM / TILE_N, 1), blk, 0, stream>>>(
          xb, Wqt + (size_t)2 * CDIM * CDIM, Vt, CDIM, CDIM, CDIM, MROWS, 0, 0, 0);

  // 2) S = scale * Q K^T (causal, upper-triangle blocks skipped)
  attn_scores_kernel<<<dim3(TSEQ / TILE_M, TSEQ / TILE_N, NBAT), blk, 0, stream>>>(
      Qb, Kb, S);

  // 3) in-place softmax -> P
  softmax_kernel<<<dim3(TSEQ / 8, NBAT), blk, 0, stream>>>(S);

  // 4) O = P @ V   (Bt = V^T, batch offset along kv dim; causal K limit)
  gemm_bf16_async<true, false, true>
      <<<dim3(TSEQ / TILE_M, CDIM / TILE_N, NBAT), blk, 0, stream>>>(
          S, Vt, Ob, TSEQ, TSEQ, MROWS, CDIM,
          (long long)TSEQ * TSEQ, (long long)TSEQ, (long long)TSEQ * CDIM);

  // 5) out = O @ Wproj  (f32 output)
  gemm_bf16_async<false, false, false>
      <<<dim3(MROWS / TILE_M, CDIM / TILE_N, 1), blk, 0, stream>>>(
          Ob, Wpt, out, CDIM, CDIM, CDIM, CDIM, 0, 0, 0);
}
